// L1RegressionActionHead_33225867002504
// MI455X (gfx1250) — matrix operand order repaired and checked by reading, not compiled, not run on hardware
//
#include <hip/hip_runtime.h>
#include <stdint.h>

// ---------------------------------------------------------------------------
// CDNA5 (gfx1250) implementation: bf16 WMMA GEMMs + fused big-LDS attention.
// wave32 only; WMMA shape 16x16x32 bf16 -> f32 accumulate.
// Global->LDS movement uses GLOBAL_LOAD_ASYNC_TO_LDS_B128 (ASYNCcnt-tracked,
// double-buffered) so tile DMA overlaps WMMA issue.
// ---------------------------------------------------------------------------

typedef __attribute__((ext_vector_type(16))) __bf16 v16bf;
typedef __attribute__((ext_vector_type(8)))  float  v8f;

#define DEVI __device__ __forceinline__

DEVI unsigned short f2bf(float f) {                 // round-to-nearest-even
  union { float f; uint32_t u; } v; v.f = f;
  uint32_t u = v.u;
  u += 0x7FFFu + ((u >> 16) & 1u);
  return (unsigned short)(u >> 16);
}
DEVI float bf2f(unsigned short h) {
  union { uint32_t u; float f; } v; v.u = ((uint32_t)h) << 16;
  return v.f;
}
DEVI uint32_t pack2(unsigned short lo, unsigned short hi) {
  return (uint32_t)lo | ((uint32_t)hi << 16);
}
union Frag { v16bf v; uint32_t u[8]; };
DEVI v8f v8zero() { v8f z = {0.f,0.f,0.f,0.f,0.f,0.f,0.f,0.f}; return z; }

// LDS byte offset of a generic pointer into shared memory: the shared aperture
// keeps the LDS offset in addr[31:0] (aperture tag lives in addr[63:32]).
DEVI uint32_t lds_ofs(const void* p) { return (uint32_t)(uintptr_t)p; }

// 16B global -> LDS async DMA (GV mode), tracked by ASYNCcnt.
DEVI void async_b128(uint32_t lds_byte_ofs, const void* gaddr) {
  asm volatile("global_load_async_to_lds_b128 %0, %1, off"
               :: "v"(lds_byte_ofs), "v"(gaddr) : "memory");
}

static constexpr int S_PAD    = 352;                // 322 keys padded to 22*16
static constexpr int ATTN_LDS = 128*S_PAD*4 + S_PAD*64*4;  // 176KB scores + 88KB K/V

// ---------------- elementwise converts -------------------------------------
__global__ void k_f2bf(const float* __restrict__ s, unsigned short* __restrict__ d, int n) {
  int i = blockIdx.x * 256 + threadIdx.x;
  if (i < n) d[i] = f2bf(s[i]);
}
__global__ void k_f2bf_strided(const float* __restrict__ s, unsigned short* __restrict__ d,
                               int nb, int rows, int cols,
                               long sbs, long soff, long dbs, long doff) {
  long i = (long)blockIdx.x * 256 + threadIdx.x;
  long tot = (long)nb * rows * cols;
  if (i >= tot) return;
  int c = (int)(i % cols);
  int r = (int)((i / cols) % rows);
  int b = (int)(i / ((long)cols * rows));
  d[doff + (long)b*dbs + (long)r*cols + c] = f2bf(s[soff + (long)b*sbs + (long)r*cols + c]);
}

// ---------------- LayerNorm (fp32 in, bf16 out) ----------------------------
__global__ __launch_bounds__(256) void k_ln(const float* __restrict__ x,
                                            const float* __restrict__ g,
                                            const float* __restrict__ bt,
                                            unsigned short* __restrict__ out, int cols) {
  __shared__ float red[256];
  int tid = threadIdx.x;
  long row = blockIdx.x;
  const float* xr = x + row * cols;
  float s = 0.f;
  for (int c = tid; c < cols; c += 256) s += xr[c];
  red[tid] = s; __syncthreads();
  for (int o = 128; o > 0; o >>= 1) { if (tid < o) red[tid] += red[tid+o]; __syncthreads(); }
  float mean = red[0] / (float)cols;
  __syncthreads();
  float v = 0.f;
  for (int c = tid; c < cols; c += 256) { float dd = xr[c] - mean; v += dd*dd; }
  red[tid] = v; __syncthreads();
  for (int o = 128; o > 0; o >>= 1) { if (tid < o) red[tid] += red[tid+o]; __syncthreads(); }
  float inv = rsqrtf(red[0] / (float)cols + 1e-5f);
  unsigned short* orow = out + row * cols;
  for (int c = tid; c < cols; c += 256) orow[c] = f2bf((xr[c] - mean) * inv * g[c] + bt[c]);
}

// ---------------- bf16 WMMA GEMM: C[M,N] = A[M,K] @ W[N,K]^T + bias --------
// EPI: 0 = store fp32; 1 = relu, store fp32 + bf16; 2 = +resid, store fp32
// Double-buffered: tile k+1 streams via async DMA while tile k runs on WMMA.
template<int EPI>
__global__ __launch_bounds__(256) void k_gemm(const unsigned short* __restrict__ A,
                                              const unsigned short* __restrict__ W,
                                              const float* __restrict__ bias,
                                              float* __restrict__ outF,
                                              unsigned short* __restrict__ outB,
                                              const float* __restrict__ resid,
                                              int N, int K) {
  __shared__ uint32_t lA[2][2048];   // 128 rows x 16 bf16-pairs (K-tile = 32)
  __shared__ uint32_t lB[2][2048];
  int tid = threadIdx.x, lane = tid & 31, wid = tid >> 5;
  int half = lane >> 4, l15 = lane & 15;
  int m0 = blockIdx.y * 128, n0 = blockIdx.x * 128;
  int wm = wid & 3, wn = wid >> 2;           // 4 row-waves x 2 col-waves
  v8f acc[2][4];
  for (int i = 0; i < 2; ++i) for (int j = 0; j < 4; ++j) acc[i][j] = v8zero();

  // each thread DMAs 4x16B per tile: 2 for A, 2 for W (issue order A,B,A,B)
  auto issueTile = [&](int kt, int buf) {
#pragma unroll
    for (int j = 0; j < 2; ++j) {
      int idx = tid + j * 256;                 // 512 quads = 128 rows x 4
      int row = idx >> 2, quad = idx & 3;      // quad = 16B (8 bf16) of the 32-elem row
      async_b128(lds_ofs(&lA[buf][row * 16 + quad * 4]),
                 A + (long)(m0 + row) * K + kt + quad * 8);
      async_b128(lds_ofs(&lB[buf][row * 16 + quad * 4]),
                 W + (long)(n0 + row) * K + kt + quad * 8);
    }
  };

  issueTile(0, 0);
  int buf = 0;
  for (int kt = 0; kt < K; kt += 32) {
    __syncthreads();                           // prior compute done before overwrite
    if (kt + 32 < K) {
      issueTile(kt + 32, buf ^ 1);
      asm volatile("s_wait_asynccnt 0x4" ::: "memory");  // tile kt landed; next 4 in flight
    } else {
      asm volatile("s_wait_asynccnt 0x0" ::: "memory");
    }
    __syncthreads();                           // tile kt visible to whole WG

    Frag fa[2];
#pragma unroll
    for (int ms = 0; ms < 2; ++ms) {
      int rb = wm * 32 + ms * 16 + l15;
#pragma unroll
      for (int v = 0; v < 8; ++v)              // A 16x32 layout: k=grp*16+half*8+2vv
        fa[ms].u[v] = lA[buf][rb * 16 + (v >> 2) * 8 + half * 4 + (v & 3)];
    }
#pragma unroll
    for (int ns = 0; ns < 4; ++ns) {
      Frag fb;
      int nb = wn * 64 + ns * 16 + l15;
#pragma unroll
      for (int v = 0; v < 8; ++v)              // B 32x16 layout: k=grp*16+half*8+2vv
        fb.u[v] = lB[buf][nb * 16 + (v >> 2) * 8 + half * 4 + (v & 3)];
#pragma unroll
      for (int ms = 0; ms < 2; ++ms)
        acc[ms][ns] = __builtin_amdgcn_wmma_f32_16x16x32_bf16(
            false, fa[ms].v, false, fb.v, (short)0, acc[ms][ns], false, false);
    }
    buf ^= 1;
  }
#pragma unroll
  for (int ms = 0; ms < 2; ++ms) {
#pragma unroll
    for (int ns = 0; ns < 4; ++ns) {
      int col = n0 + wn * 64 + ns * 16 + l15;
      float bv = bias[col];
#pragma unroll
      for (int r = 0; r < 8; ++r) {            // C layout: M = r + half*8
        int row = m0 + wm * 32 + ms * 16 + r + half * 8;
        long o = (long)row * N + col;
        float val = acc[ms][ns][r] + bv;
        if (EPI == 1) { val = fmaxf(val, 0.f); outF[o] = val; outB[o] = f2bf(val); }
        else if (EPI == 2) { outF[o] = val + resid[o]; }
        else { outF[o] = val; }
      }
    }
  }
}

// ---------------- RoPE: fp32 q/k -> bf16 q-buf + k_all[s=t] ----------------
__global__ void k_rope(const float* __restrict__ q, const float* __restrict__ k,
                       uint32_t* __restrict__ qb, uint32_t* __restrict__ kall) {
  int idx = blockIdx.x * 256 + threadIdx.x;
  int p = idx & 63; int r = idx >> 6;
  int h = r & 7; r >>= 3;
  int t = r & 255; int b = r >> 8;
  if (b >= 64) return;
  long base = ((long)(b * 256 + t)) * 1024 + h * 128 + 2 * p;
  float q0 = q[base], q1 = q[base + 1];
  float k0 = k[base], k1 = k[base + 1];
  float f0 = __powf(10000.f, -(float)((2 * p) & 63) * (1.f / 64.f));
  float f1 = __powf(10000.f, -(float)((2 * p + 1) & 63) * (1.f / 64.f));
  float a0 = (float)t * f0, a1 = (float)t * f1;
  float s0, c0, s1, c1;
  __sincosf(a0, &s0, &c0); __sincosf(a1, &s1, &c1);
  uint32_t qw = pack2(f2bf(q0 * c0 - q1 * s0), f2bf(q1 * c1 + q0 * s1));
  uint32_t kw = pack2(f2bf(k0 * c0 - k1 * s0), f2bf(k1 * c1 + k0 * s1));
  int bh = b * 8 + h;
  qb[((long)bh * 256 + t) * 64 + p] = qw;
  kall[((long)bh * S_PAD + t) * 64 + p] = kw;
}

// ---------------- scatter (B*rowsPerB,1024) fp32 -> k_all/v_all bf16 -------
__global__ void k_scatter_kv(const float* __restrict__ src, unsigned short* __restrict__ dst,
                             int rowsPerB, int sOff) {
  long i = (long)blockIdx.x * 256 + threadIdx.x;
  long tot = (long)64 * rowsPerB * 1024;
  if (i >= tot) return;
  int col = (int)(i & 1023);
  long row = i >> 10;
  int b = (int)(row / rowsPerB);
  int r = (int)(row % rowsPerB);
  int h = col >> 7, d = col & 127;
  dst[((long)(b * 8 + h) * S_PAD + sOff + r) * 128 + d] = f2bf(src[i]);
}

// ---------------- fused attention: scores->softmax->out, all WMMA ----------
__global__ __launch_bounds__(256) void k_attn(const unsigned short* __restrict__ qbu,
                                              const uint32_t* __restrict__ kallp,
                                              const unsigned short* __restrict__ vall,
                                              const float* __restrict__ gate, int layer,
                                              float* __restrict__ out) {
  extern __shared__ char smem[];
  float*    sc = (float*)smem;                          // 128 x 352 fp32 scores
  uint32_t* kv = (uint32_t*)(smem + 128 * S_PAD * 4);   // 352x64 K-pairs, then 128x176 V^T
  int tid = threadIdx.x, lane = tid & 31, wid = tid >> 5;
  int half = lane >> 4, m = lane & 15;
  int bid = blockIdx.x;
  int tblk = bid & 1, bh = bid >> 1;
  int b = bh >> 3, h = bh & 7;
  int t0 = tblk * 128;

  // async-DMA K tile (88KB contiguous) straight into LDS: 22 x b128 per thread
  const uint32_t* kp = kallp + (long)bh * S_PAD * 64;
#pragma unroll
  for (int j = 0; j < 22; ++j) {
    int idx = tid + j * 256;                   // quad index (4 words = 16B)
    async_b128(lds_ofs(&kv[idx * 4]), kp + idx * 4);
  }

  // build the 4 q A-fragments (K=128 split into 4 WMMA K-steps) from global
  const uint32_t* qp = (const uint32_t*)qbu + ((long)bh * 256 + t0 + wid * 16) * 64;
  Frag aq[4];
#pragma unroll
  for (int kk = 0; kk < 4; ++kk)
#pragma unroll
    for (int v = 0; v < 8; ++v)
      aq[kk].u[v] = qp[m * 64 + kk * 16 + (v >> 2) * 8 + half * 4 + (v & 3)];

  asm volatile("s_wait_asynccnt 0x0" ::: "memory");
  __syncthreads();

  float gv = tanhf(gate[layer]);
  const float scale = 0.08838834764831845f;   // 1/sqrt(128)
  for (int nt = 0; nt < 22; ++nt) {           // 22 key tiles of 16
    v8f acc = v8zero();
#pragma unroll
    for (int kk = 0; kk < 4; ++kk) {
      Frag bk;
#pragma unroll
      for (int v = 0; v < 8; ++v)
        bk.u[v] = kv[(nt * 16 + m) * 64 + kk * 16 + (v >> 2) * 8 + half * 4 + (v & 3)];
      acc = __builtin_amdgcn_wmma_f32_16x16x32_bf16(
          false, aq[kk].v, false, bk.v, (short)0, acc, false, false);
    }
    int col = nt * 16 + m;
    float f = scale;
    if (col >= 258 && col < 322) f *= gv;     // task keys get tanh(gate)
#pragma unroll
    for (int r = 0; r < 8; ++r)
      sc[(wid * 16 + r + half * 8) * S_PAD + col] = acc[r] * f;
  }
  __syncthreads();

  // softmax over 322 valid keys, one thread per row; zero the pad
  if (tid < 128) {
    float* row = sc + (long)tid * S_PAD;
    float mx = -1e30f;
    for (int s = 0; s < 322; ++s) mx = fmaxf(mx, row[s]);
    float sum = 0.f;
    for (int s = 0; s < 322; ++s) { float e = __expf(row[s] - mx); row[s] = e; sum += e; }
    float inv = 1.f / sum;
    for (int s = 0; s < 322; ++s) row[s] *= inv;
    for (int s = 322; s < S_PAD; ++s) row[s] = 0.f;
  }
  __syncthreads();

  // load V transposed into same LDS region: kv[d][s-pair] (needs pack, VALU path)
  const unsigned short* vp = vall + (long)bh * S_PAD * 128;
  for (int i = tid; i < 128 * 176; i += 256) {
    int d = i / 176, sp = i % 176;
    kv[d * 176 + sp] = pack2(vp[(long)(2 * sp) * 128 + d], vp[(long)(2 * sp + 1) * 128 + d]);
  }
  __syncthreads();

  v8f o[8];
#pragma unroll
  for (int j = 0; j < 8; ++j) o[j] = v8zero();
  for (int st = 0; st < 11; ++st) {           // S_PAD = 11 WMMA K-steps of 32
    Frag ap;
    const float* prow = sc + (long)(wid * 16 + m) * S_PAD;
#pragma unroll
    for (int v = 0; v < 8; ++v) {
      int k = st * 32 + (v >> 2) * 16 + half * 8 + (v & 3) * 2;
      ap.u[v] = pack2(f2bf(prow[k]), f2bf(prow[k + 1]));
    }
#pragma unroll
    for (int dt = 0; dt < 8; ++dt) {
      Frag bv;
#pragma unroll
      for (int v = 0; v < 8; ++v)
        bv.u[v] = kv[(dt * 16 + m) * 176 + st * 16 + (v >> 2) * 8 + half * 4 + (v & 3)];
      o[dt] = __builtin_amdgcn_wmma_f32_16x16x32_bf16(
          false, ap.v, false, bv.v, (short)0, o[dt], false, false);
    }
  }
  float* op = out + ((long)(b * 256) + t0 + wid * 16) * 1024 + h * 128;
#pragma unroll
  for (int dt = 0; dt < 8; ++dt)
#pragma unroll
    for (int r = 0; r < 8; ++r)
      op[(long)(r + half * 8) * 1024 + dt * 16 + m] = o[dt][r];
}

// ---------------- final 1024 -> 12 head ------------------------------------
__global__ void k_out(const unsigned short* __restrict__ a, const float* __restrict__ W2,
                      const float* __restrict__ b2, float* __restrict__ out, int rows) {
  int i = blockIdx.x * 256 + threadIdx.x;
  if (i >= rows * 12) return;
  int r = i / 12, o = i % 12;
  const unsigned short* ar = a + (long)r * 1024;
  const float* w = W2 + o * 1024;
  float s = b2[o];
  for (int c = 0; c < 1024; ++c) s += bf2f(ar[c]) * w[c];
  out[i] = s;
}

// ---------------------------------------------------------------------------
extern "C" void kernel_launch(void* const* d_in, const int* in_sizes, int n_in,
                              void* d_out, int out_size, void* d_ws, size_t ws_size,
                              hipStream_t stream) {
  (void)in_sizes; (void)n_in; (void)out_size; (void)ws_size;
  const float* x    = (const float*)d_in[0];
  const float* h_a  = (const float*)d_in[1];
  const float* h_t  = (const float*)d_in[2];
  const float* p    = (const float*)d_in[3];
  const float* ln1g = (const float*)d_in[4];
  const float* ln1b = (const float*)d_in[5];
  const float* W1   = (const float*)d_in[6];
  const float* b1   = (const float*)d_in[7];
  const float* Wq   = (const float*)d_in[8];
  const float* bq   = (const float*)d_in[9];
  const float* Wks  = (const float*)d_in[10];
  const float* bks  = (const float*)d_in[11];
  const float* Wvs  = (const float*)d_in[12];
  const float* bvs  = (const float*)d_in[13];
  const float* Wka  = (const float*)d_in[14];
  const float* bka  = (const float*)d_in[15];
  const float* Wva  = (const float*)d_in[16];
  const float* bva  = (const float*)d_in[17];
  const float* Wkt  = (const float*)d_in[18];
  const float* bkt  = (const float*)d_in[19];
  const float* Wvt  = (const float*)d_in[20];
  const float* bvt  = (const float*)d_in[21];
  const float* Wo   = (const float*)d_in[22];
  const float* bo   = (const float*)d_in[23];
  const float* gate = (const float*)d_in[24];
  const float* flng = (const float*)d_in[25];
  const float* flnb = (const float*)d_in[26];
  const float* Wf   = (const float*)d_in[27];
  const float* bfb  = (const float*)d_in[28];
  const float* ln2g = (const float*)d_in[29];
  const float* ln2b = (const float*)d_in[30];
  const float* W2   = (const float*)d_in[31];
  const float* b2   = (const float*)d_in[32];

  char* base = (char*)d_ws;
  size_t off = 0;
  auto alloc = [&](size_t bytes) -> void* {
    void* pp = base + off;
    off += (bytes + 255) & ~(size_t)255;
    return pp;
  };
  float* hf = (float*)alloc(16384L * 1024 * 4);   // current hidden, fp32
  float* t0 = (float*)alloc(16384L * 1024 * 4);
  float* t1 = (float*)alloc(16384L * 1024 * 4);
  float* t2 = (float*)alloc(16384L * 1024 * 4);
  unsigned short* hb   = (unsigned short*)alloc(16384L * 1024 * 2);  // hidden bf16
  unsigned short* ab   = (unsigned short*)alloc(16384L * 4096 * 2);  // bf16 act scratch
  unsigned short* qb   = (unsigned short*)alloc(512L * 256 * 128 * 2);
  unsigned short* kall = (unsigned short*)alloc(512L * S_PAD * 128 * 2);
  unsigned short* vall = (unsigned short*)alloc(512L * S_PAD * 128 * 2);
  unsigned short* wb   = (unsigned short*)alloc((4194304L + 18L * 1048576) * 2);
  unsigned short* adb  = (unsigned short*)alloc(128L * 1024 * 2);
  unsigned short* htb  = (unsigned short*)alloc(4096L * 1024 * 2);

  unsigned short* wbW1 = wb;
  auto wbl = [&](int i, int j) { return wb + 4194304L + ((long)i * 9 + j) * 1048576L; };

  hipMemsetAsync(kall, 0, 512L * S_PAD * 128 * 2, stream);
  hipMemsetAsync(vall, 0, 512L * S_PAD * 128 * 2, stream);

  // convert weights to bf16
  k_f2bf<<<(4194304 + 255) / 256, 256, 0, stream>>>(W1, wbW1, 4194304);
  const float* wsrc[9] = {Wq, Wks, Wvs, Wka, Wva, Wkt, Wvt, Wo, Wf};
  for (int i = 0; i < 2; ++i)
    for (int j = 0; j < 9; ++j)
      k_f2bf<<<(1048576 + 255) / 256, 256, 0, stream>>>(wsrc[j] + (long)i * 1048576, wbl(i, j), 1048576);

  hipFuncSetAttribute(reinterpret_cast<const void*>(k_attn),
                      hipFuncAttributeMaxDynamicSharedMemorySize, ATTN_LDS);

  // stem: LN1 then W1 GEMM (+relu, dual fp32/bf16 store)
  k_ln<<<16384, 256, 0, stream>>>(x, ln1g, ln1b, ab, 4096);
  k_gemm<1><<<dim3(8, 128), 256, 0, stream>>>(ab, wbW1, b1, hf, hb, nullptr, 1024, 4096);

  for (int i = 0; i < 2; ++i) {
    long wo = (long)i * 1024;   // per-layer bias offset
    // q, k_self projections -> RoPE -> bf16 q / k_all[s<256]
    k_gemm<0><<<dim3(8, 128), 256, 0, stream>>>(hb, wbl(i, 0), bq + wo,  t0, nullptr, nullptr, 1024, 1024);
    k_gemm<0><<<dim3(8, 128), 256, 0, stream>>>(hb, wbl(i, 1), bks + wo, t1, nullptr, nullptr, 1024, 1024);
    k_rope<<<8388608 / 256, 256, 0, stream>>>(t0, t1, (uint32_t*)qb, (uint32_t*)kall);
    // v_self
    k_gemm<0><<<dim3(8, 128), 256, 0, stream>>>(hb, wbl(i, 2), bvs + wo, t2, nullptr, nullptr, 1024, 1024);
    k_scatter_kv<<<(16777216 + 255) / 256, 256, 0, stream>>>(t2, vall, 256, 0);
    // adapter keys/values: h_ad = concat(h_a[:,i], p) -> bf16 (B*2,1024)
    k_f2bf_strided<<<(65536 + 255) / 256, 256, 0, stream>>>(h_a, adb, 64, 1, 1024, 2048, (long)i * 1024, 2048, 0);
    k_f2bf_strided<<<(65536 + 255) / 256, 256, 0, stream>>>(p,   adb, 64, 1, 1024, 1024, 0,              2048, 1024);
    k_gemm<0><<<dim3(8, 1), 256, 0, stream>>>(adb, wbl(i, 3), bka + wo, t0, nullptr, nullptr, 1024, 1024);
    k_scatter_kv<<<(131072 + 255) / 256, 256, 0, stream>>>(t0, kall, 2, 256);
    k_gemm<0><<<dim3(8, 1), 256, 0, stream>>>(adb, wbl(i, 4), bva + wo, t0, nullptr, nullptr, 1024, 1024);
    k_scatter_kv<<<(131072 + 255) / 256, 256, 0, stream>>>(t0, vall, 2, 256);
    // task keys/values: h_t[:,i] (B*64,1024)
    k_f2bf_strided<<<(4194304 + 255) / 256, 256, 0, stream>>>(h_t, htb, 64, 64, 1024, 131072, (long)i * 65536, 65536, 0);
    k_gemm<0><<<dim3(8, 32), 256, 0, stream>>>(htb, wbl(i, 5), bkt + wo, t0, nullptr, nullptr, 1024, 1024);
    k_scatter_kv<<<(4194304 + 255) / 256, 256, 0, stream>>>(t0, kall, 64, 258);
    k_gemm<0><<<dim3(8, 32), 256, 0, stream>>>(htb, wbl(i, 6), bvt + wo, t0, nullptr, nullptr, 1024, 1024);
    k_scatter_kv<<<(4194304 + 255) / 256, 256, 0, stream>>>(t0, vall, 64, 258);
    // fused attention (264KB LDS/WG) -> t2 fp32
    k_attn<<<1024, 256, ATTN_LDS, stream>>>(qb, (const uint32_t*)kall, vall, gate, i, t2);
    // Wo projection + residual(hf) -> t0
    k_f2bf<<<(16777216 + 255) / 256, 256, 0, stream>>>(t2, ab, 16777216);
    k_gemm<2><<<dim3(8, 128), 256, 0, stream>>>(ab, wbl(i, 7), bo + wo, t0, nullptr, hf, 1024, 1024);
    // post-LN then FFN projection + relu -> new hidden
    k_ln<<<16384, 256, 0, stream>>>(t0, flng + wo, flnb + wo, ab, 1024);
    k_gemm<1><<<dim3(8, 128), 256, 0, stream>>>(ab, wbl(i, 8), bfb + wo, hf, hb, nullptr, 1024, 1024);
  }

  // final LN + 1024->12 head
  k_ln<<<16384, 256, 0, stream>>>(hf, ln2g, ln2b, ab, 1024);
  k_out<<<(16384 * 12 + 255) / 256, 256, 0, stream>>>(ab, W2, b2, (float*)d_out, 16384);
}
